// MyModel_31877247271123
// MI455X (gfx1250) — matrix operand order repaired and checked by reading, not compiled, not run on hardware
//
#include <hip/hip_runtime.h>
#include <cstdint>
#include <cstddef>

// ---------------------------------------------------------------------------
// Types for CDNA5 WMMA (gfx1250, wave32)
// ---------------------------------------------------------------------------
typedef __bf16 bf16_t;
typedef __attribute__((ext_vector_type(16))) __bf16 v16bf;
typedef __attribute__((ext_vector_type(8)))  __bf16 v8bf;
typedef __attribute__((ext_vector_type(8)))  float  v8f;
typedef __attribute__((ext_vector_type(4)))  unsigned int v4u;
typedef __attribute__((ext_vector_type(8)))  int v8i;
typedef __attribute__((ext_vector_type(4)))  int v4i;

__device__ __forceinline__ float  bf2f(bf16_t v) { return (float)v; }
__device__ __forceinline__ bf16_t f2bf(float v)  { return (bf16_t)v; }

#define CONCAT16(lo, hi) \
  __builtin_shufflevector(lo, hi, 0, 1, 2, 3, 4, 5, 6, 7, 8, 9, 10, 11, 12, 13, 14, 15)

#define WMMA_BF16(a, b, c) \
  __builtin_amdgcn_wmma_f32_16x16x32_bf16(false, a, false, b, (short)0, c, false, false)

// Tensor Data Mover availability (this toolchain: 6-arg builtin,
// (uint32x4 g0, int32x8 g1, int32x4 g2, int32x4 g3, int32x8 g4, i32 cpol))
#if defined(__HIP_DEVICE_COMPILE__) && defined(__has_builtin)
#if __has_builtin(__builtin_amdgcn_tensor_load_to_lds) && \
    __has_builtin(__builtin_amdgcn_s_wait_tensorcnt)
#define FRCNN_TDM 1
#endif
#endif
#ifndef FRCNN_TDM
#define FRCNN_TDM 0
#endif

// ---------------------------------------------------------------------------
// Model constants
// ---------------------------------------------------------------------------
#define IMG 256
#define BATCH 8
#define FEAT_HW 32
#define FEAT_CH 128
#define NUM_ANCHOR 21
#define NA_TOT (FEAT_HW * FEAT_HW * NUM_ANCHOR) // 21504
#define MAX_CAND 32
#define POOLP 7
#define POOLK (FEAT_CH * POOLP * POOLP)         // 6272
#define NKT 8                                    // K-chunks staged per TDM op

// ---------------------------------------------------------------------------
// Layout transform: x NCHW f32 -> NHWC bf16
// ---------------------------------------------------------------------------
__global__ void frcnn_nchw2nhwc(const float* __restrict__ x, bf16_t* __restrict__ out,
                                int B, int C, int H, int W) {
  long i = (long)blockIdx.x * blockDim.x + threadIdx.x;
  long tot = (long)B * C * H * W;
  if (i >= tot) return;
  int w = i % W; long r = i / W;
  int h = r % H; r /= H;
  int c = r % C; int b = r / C;
  out[(((long)b * H + h) * W + w) * C + c] = f2bf(x[i]);
}

// ---------------------------------------------------------------------------
// Weight repack into WMMA-fragment-swizzled layout:
//   wp[ktile][ntile][lane][e]  (16 bf16 contiguous per lane = one B fragment)
// Conv K-space is padded-channel tap-major: k = (ky*ks+kx)*Cpad + ci.
// Fragment mapping (B 32x16): n = ntile*16 + lane%16,
//   k = ktile*32 + (e<8 ? 8*(lane/16) + e : 16 + 8*(lane/16) + e - 8).
// ---------------------------------------------------------------------------
__global__ void frcnn_pack_conv(const float* __restrict__ w, int Cout, int Cin,
                                int Cpad, int ks, bf16_t* __restrict__ wp, int Np) {
  long i = (long)blockIdx.x * blockDim.x + threadIdx.x;
  int ntiles = Np >> 4;
  long tot = (long)(ks * ks * Cpad / 32) * ntiles * 512;
  if (i >= tot) return;
  int e = i & 15;
  int lane = (i >> 4) & 31;
  long r = i >> 9;
  int tn = r % ntiles;
  int ktile = (int)(r / ntiles);
  int n = tn * 16 + (lane & 15);
  int khalf = (lane < 16) ? 0 : 8;
  int k = ktile * 32 + khalf + ((e < 8) ? e : (16 + e - 8));
  int ci = k % Cpad, tap = k / Cpad;
  int kx = tap % ks, ky = tap / ks;
  float v = 0.f;
  if (n < Cout && ci < Cin)
    v = w[(((long)n * Cin + ci) * ks + ky) * ks + kx];
  wp[i] = f2bf(v);
}

// linear w [N,K] f32 -> fragment-swizzled (K % 32 == 0)
__global__ void frcnn_pack_lin(const float* __restrict__ w, int N, int K,
                               bf16_t* __restrict__ wp, int Np) {
  long i = (long)blockIdx.x * blockDim.x + threadIdx.x;
  int ntiles = Np >> 4;
  long tot = (long)(K / 32) * ntiles * 512;
  if (i >= tot) return;
  int e = i & 15;
  int lane = (i >> 4) & 31;
  long r = i >> 9;
  int tn = r % ntiles;
  int ktile = (int)(r / ntiles);
  int n = tn * 16 + (lane & 15);
  int k = ktile * 32 + ((lane < 16) ? 0 : 8) + ((e < 8) ? e : (16 + e - 8));
  wp[i] = f2bf(n < N ? w[(long)n * K + k] : 0.f);
}

// ---------------------------------------------------------------------------
// Fused conv + BN(scale,bias) [+residual] [+ReLU] as implicit GEMM via WMMA.
// One wave per 16(pixel) x 32(Cout) tile: A fragment reused across 2 B tiles.
// Fast path (Cin%32==0): A fragment = two aligned 16B vector loads.
// ---------------------------------------------------------------------------
__global__ __launch_bounds__(32)
void frcnn_conv_wmma(const bf16_t* __restrict__ in, int Cin, int Cpad,
                     int Hin, int Win,
                     const bf16_t* __restrict__ wp, int Np,
                     const float* __restrict__ scale, const float* __restrict__ bias,
                     const bf16_t* __restrict__ residual,
                     bf16_t* __restrict__ out, int Cout, int Hout, int Wout,
                     int ks, int stride, int pad, int doRelu) {
  const int lane = threadIdx.x;
  const int tm = blockIdx.x, tn2 = blockIdx.y, b = blockIdx.z;
  const int npix = Hout * Wout;
  const int mlo = lane & 15;
  const bool hi = lane >= 16;
  const int kbase = hi ? 8 : 0;
  const int ntilesN = Np >> 4;

  const int m = tm * 16 + mlo;
  const bool mvalid = m < npix;
  const int oy = mvalid ? m / Wout : 0;
  const int ox = mvalid ? m % Wout : 0;
  const bf16_t* inB = in + (long)b * Hin * Win * Cin;
  const bool vecCin = (Cin & 31) == 0;

  v8f acc0 = {}, acc1 = {};
  int ktile = 0;
  for (int ky = 0; ky < ks; ++ky) {
    const int iy = oy * stride + ky - pad;
    const bool yok = mvalid && (unsigned)iy < (unsigned)Hin;
    for (int kx = 0; kx < ks; ++kx) {
      const int ix = ox * stride + kx - pad;
      const bool ok = yok && (unsigned)ix < (unsigned)Win;
      const bf16_t* row = inB + ((long)iy * Win + ix) * Cin;
      for (int cc = 0; cc < Cpad; cc += 32, ++ktile) {
        v16bf a;
        if (vecCin) {
          v8bf lo = {}, hi8 = {};
          if (ok) {
            lo  = *(const v8bf*)(row + cc + kbase);
            hi8 = *(const v8bf*)(row + cc + 16 + kbase);
          }
          a = CONCAT16(lo, hi8);
        } else {
#pragma unroll
          for (int e = 0; e < 16; ++e) {
            int koff = (e < 8) ? (kbase + e) : (16 + kbase + (e - 8));
            int ci = cc + koff;
            a[e] = (ok && ci < Cin) ? row[ci] : f2bf(0.f);
          }
        }
        const bf16_t* bptr = wp + (((long)ktile * ntilesN + tn2 * 2) * 32 + lane) * 16;
        v16bf bw0 = *(const v16bf*)bptr;
        v16bf bw1 = *(const v16bf*)(bptr + 512);
        __builtin_prefetch(bptr + (long)ntilesN * 512, 0, 1);
        acc0 = WMMA_BF16(a, bw0, acc0);
        acc1 = WMMA_BF16(a, bw1, acc1);
      }
    }
  }

#pragma unroll
  for (int t = 0; t < 2; ++t) {
    int n = tn2 * 32 + t * 16 + mlo;
    if (n < Cout) {
      float sc = scale[n], bi = bias[n];
      v8f acc = t ? acc1 : acc0;
#pragma unroll
      for (int r = 0; r < 8; ++r) {
        int mm = tm * 16 + r + (hi ? 8 : 0);
        if (mm < npix) {
          long oidx = ((long)b * npix + mm) * Cout + n;
          float v = acc[r] * sc + bi;
          if (residual) v += bf2f(residual[oidx]);
          if (doRelu) v = fmaxf(v, 0.f);
          out[oidx] = f2bf(v);
        }
      }
    }
  }
}

// ---------------------------------------------------------------------------
// GEMM: out = [relu](A[M,K]bf16 x W + bias). K%32==0, M%16==0, Np%32==0.
// One wave per 16M x 32N tile. B operand staged through LDS by the Tensor
// Data Mover (2-D strided tile of the swizzled weight buffer), fenced with
// s_wait_tensorcnt; falls back to direct global loads without TDM support.
// ---------------------------------------------------------------------------
__global__ __launch_bounds__(32)
void frcnn_gemm_wmma(const bf16_t* __restrict__ A, int M, int K,
                     const bf16_t* __restrict__ wp, int Np,
                     const float* __restrict__ bias, int N,
                     float* __restrict__ outF, bf16_t* __restrict__ outB, int doRelu) {
  const int lane = threadIdx.x;
  const int tm = blockIdx.x, tn2 = blockIdx.y;
  const int mlo = lane & 15;
  const bool hi = lane >= 16;
  const int kbase = hi ? 8 : 0;
  const int ntilesN = Np >> 4;
  const int m = tm * 16 + mlo;
  const bf16_t* arow = A + (long)m * K;

  v8f acc0 = {}, acc1 = {};
  const int nkt = K >> 5;

#if FRCNN_TDM
  __shared__ bf16_t lb[NKT * 1024];     // NKT rows x (2 fragments x 512 bf16)
  const uint32_t lds_base = (uint32_t)(uintptr_t)lb;
  int ktile = 0;
  for (int kt0 = 0; kt0 < nkt; kt0 += NKT) {
    const int rows = (nkt - kt0 < NKT) ? (nkt - kt0) : NKT;
    // previous stage's LDS reads must retire before TDM overwrites the buffer
    asm volatile("s_wait_dscnt 0x0" ::: "memory");
    // ---- D# descriptor: 2-D tile, rows of 2KB, row stride = ntilesN KB ----
    const uint64_t gaddr =
        (uint64_t)(uintptr_t)(wp + ((size_t)kt0 * ntilesN + (size_t)tn2 * 2) * 512);
    v4u g0;
    g0.x = 1u;                                    // count=1, user descriptor
    g0.y = lds_base;                              // LDS byte address
    g0.z = (uint32_t)gaddr;                       // global_addr[31:0]
    g0.w = (uint32_t)((gaddr >> 32) & 0x01FFFFFFu) | (2u << 30); // addr[56:32], type=2
    const uint32_t rowUnits = 256u;               // 2KB per row in 8B units
    const uint32_t strideU  = (uint32_t)ntilesN * 128u; // row stride in 8B units
    v8i g1;
    g1[0] = (int)(3u << 16);                      // data_size=3 (8 bytes)
    g1[1] = (int)((strideU & 0xFFFFu) << 16);     // tensor_dim0[15:0]
    g1[2] = (int)(((strideU >> 16) & 0xFFFFu) |   // tensor_dim0[31:16]
                  (((uint32_t)rows & 0xFFFFu) << 16)); // tensor_dim1[15:0]
    g1[3] = (int)(rowUnits << 16);                // tensor_dim1[31:16]=0, tile_dim0
    g1[4] = (int)((uint32_t)rows & 0xFFFFu);      // tile_dim1 = rows, tile_dim2 = 0
    g1[5] = (int)strideU;                         // tensor_dim0_stride[31:0]
    g1[6] = 0;                                    // stride hi, dim1 stride lo
    g1[7] = 0;
    v4i g2 = {0, 0, 0, 0}, g3 = {0, 0, 0, 0};
    v8i g4 = {0, 0, 0, 0, 0, 0, 0, 0};
    __builtin_amdgcn_tensor_load_to_lds(g0, g1, g2, g3, g4, 0);
    __builtin_amdgcn_s_wait_tensorcnt((short)0);

    for (int j = 0; j < rows; ++j, ++ktile) {
      const int kc = ktile * 32;
      v8bf lo  = *(const v8bf*)(arow + kc + kbase);
      v8bf hi8 = *(const v8bf*)(arow + kc + 16 + kbase);
      v16bf a = CONCAT16(lo, hi8);
      v16bf bw0 = *(const v16bf*)(lb + (size_t)j * 1024 + lane * 16);
      v16bf bw1 = *(const v16bf*)(lb + (size_t)j * 1024 + 512 + lane * 16);
      acc0 = WMMA_BF16(a, bw0, acc0);
      acc1 = WMMA_BF16(a, bw1, acc1);
    }
  }
#else
  for (int ktile = 0; ktile < nkt; ++ktile) {
    const int kc = ktile * 32;
    v8bf lo  = *(const v8bf*)(arow + kc + kbase);
    v8bf hi8 = *(const v8bf*)(arow + kc + 16 + kbase);
    v16bf a = CONCAT16(lo, hi8);
    const bf16_t* bptr = wp + (((long)ktile * ntilesN + tn2 * 2) * 32 + lane) * 16;
    v16bf bw0 = *(const v16bf*)bptr;
    v16bf bw1 = *(const v16bf*)(bptr + 512);
    acc0 = WMMA_BF16(a, bw0, acc0);
    acc1 = WMMA_BF16(a, bw1, acc1);
  }
#endif

#pragma unroll
  for (int t = 0; t < 2; ++t) {
    int n = tn2 * 32 + t * 16 + mlo;
    float bi = (n < N) ? bias[n] : 0.f;
    v8f acc = t ? acc1 : acc0;
#pragma unroll
    for (int r = 0; r < 8; ++r) {
      int mm = tm * 16 + r + (hi ? 8 : 0);
      if (mm < M && n < N) {
        float v = acc[r] + bi;
        if (doRelu) v = fmaxf(v, 0.f);
        if (outF) outF[(long)mm * N + n] = v;
        if (outB) outB[(long)mm * N + n] = f2bf(v);
      }
    }
  }
}

// ---------------------------------------------------------------------------
// Top-32 anchor selection (softmax class-1 prob ~ logit diff; jax tie-break:
// lower index wins on equal value) + anchor regen + box adjust.
// ---------------------------------------------------------------------------
__global__ __launch_bounds__(256)
void frcnn_topk_boxes(const float* __restrict__ labels, const float* __restrict__ offs,
                      float* __restrict__ boxes_out, float* __restrict__ boxes_ws) {
  const int b = blockIdx.x, tid = threadIdx.x;
  __shared__ float sv[256];
  __shared__ int si[256];
  __shared__ int chosen[MAX_CAND];
  const float* lab = labels + (long)b * NA_TOT * 2;

  for (int sel = 0; sel < MAX_CAND; ++sel) {
    float best = -3.4e38f; int bidx = 0x7fffffff;
    for (int i = tid; i < NA_TOT; i += 256) {
      bool taken = false;
      for (int j = 0; j < sel; ++j) taken |= (chosen[j] == i);
      if (taken) continue;
      float s = lab[i * 2 + 1] - lab[i * 2 + 0];
      if (s > best || (s == best && i < bidx)) { best = s; bidx = i; }
    }
    sv[tid] = best; si[tid] = bidx;
    __syncthreads();
    for (int off = 128; off > 0; off >>= 1) {
      if (tid < off) {
        float ov = sv[tid + off]; int oi = si[tid + off];
        if (ov > sv[tid] || (ov == sv[tid] && oi < si[tid])) { sv[tid] = ov; si[tid] = oi; }
      }
      __syncthreads();
    }
    if (tid == 0) chosen[sel] = si[0];
    __syncthreads();
  }

  if (tid < MAX_CAND) {
    int i = chosen[tid];
    int a = i % NUM_ANCHOR, cell = i / NUM_ANCHOR;
    int ycell = cell % FEAT_HW, xcell = cell / FEAT_HW;
    float xc = xcell * 8.f + 4.f, yc = ycell * 8.f + 4.f;
    const float SC[7] = {0.5f, 1.f, 2.f, 3.f, 4.f, 5.f, 6.f};
    const int RX[3] = {1, 1, 2}, RY[3] = {2, 1, 1};
    float ww = 8.f * SC[a / 3] * RX[a % 3];
    float hh = 8.f * SC[a / 3] * RY[a % 3];
    int xx = (int)(xc - ww * 0.5f); if (xx < 0) xx = 0;
    int yy = (int)(yc - hh * 0.5f); if (yy < 0) yy = 0;
    float bw = fminf((float)(int)ww, (float)(IMG - xx));
    float bh = fminf((float)(int)hh, (float)(IMG - yy));
    const float* o = offs + ((long)b * NA_TOT + i) * 4;
    float nx = xx + bw * o[0], ny = yy + bh * o[1];
    float nw = bw * expf(o[2]), nh = bh * expf(o[3]);
    long base = ((long)b * MAX_CAND + tid) * 4;
    boxes_out[base + 0] = nx; boxes_out[base + 1] = ny;
    boxes_out[base + 2] = nw; boxes_out[base + 3] = nh;
    boxes_ws[base + 0] = nx;  boxes_ws[base + 1] = ny;
    boxes_ws[base + 2] = nw;  boxes_ws[base + 3] = nh;
  }
}

// ---------------------------------------------------------------------------
// ROI bilinear crop (reference's swapped-axis affine grid), feat NHWC bf16.
// pooled row layout: k = c*49 + py*7 + px  -> [B*R, 6272] bf16.
// ---------------------------------------------------------------------------
__global__ __launch_bounds__(256)
void frcnn_roi_pool(const bf16_t* __restrict__ feat, const float* __restrict__ boxes,
                    bf16_t* __restrict__ pooled) {
  const int br = blockIdx.x;            // b*32 + r
  const int b = br >> 5;
  const float* bx = boxes + (long)br * 4;
  float x1 = bx[0], y1 = bx[1], x2 = x1 + bx[2], y2 = y1 + bx[3];
  float ax = (y2 - y1) / (float)IMG, cx = (y2 + y1) / (float)IMG - 1.f;
  float by = (x2 - x1) / (float)IMG, dy = (x2 + x1) / (float)IMG - 1.f;
  const bf16_t* fb = feat + (long)b * FEAT_HW * FEAT_HW * FEAT_CH;

  for (int idx = threadIdx.x; idx < POOLK; idx += 256) {
    int c = idx / 49, p = idx % 49, py = p / 7, px = p % 7;
    float basex = (2.f * px + 1.f) / 7.f - 1.f;
    float basey = (2.f * py + 1.f) / 7.f - 1.f;
    float x_in = basex * ax + cx;
    float y_in = basey * by + dy;
    float ix = ((x_in + 1.f) * FEAT_HW - 1.f) * 0.5f;
    float iy = ((y_in + 1.f) * FEAT_HW - 1.f) * 0.5f;
    float ix0 = floorf(ix), iy0 = floorf(iy);
    float wx = ix - ix0, wy = iy - iy0;
    int xi = (int)ix0, yi = (int)iy0;
    auto g = [&](int yy, int xx) -> float {
      if ((unsigned)yy < (unsigned)FEAT_HW && (unsigned)xx < (unsigned)FEAT_HW)
        return bf2f(fb[((long)yy * FEAT_HW + xx) * FEAT_CH + c]);
      return 0.f;
    };
    float v = g(yi, xi) * (1.f - wx) * (1.f - wy)
            + g(yi, xi + 1) * wx * (1.f - wy)
            + g(yi + 1, xi) * (1.f - wx) * wy
            + g(yi + 1, xi + 1) * wx * wy;
    pooled[(long)br * POOLK + idx] = f2bf(v);
  }
}

// ---------------------------------------------------------------------------
// Host orchestration
// ---------------------------------------------------------------------------
struct ConvP { const float *bias, *scale, *w; };
struct LinP  { const float *b,   *w; };
struct BackboneP { ConvP stem; ConvP c1[8], c2[8], idn[8]; };

static const int BCFG[8][3] = {{4,16,1},{16,16,1},{16,32,2},{32,32,1},
                               {32,64,2},{64,64,1},{64,128,2},{128,128,1}};
static const int HASIDN[8] = {1,0,1,0,1,0,1,0};

static int parse_backbone(void* const* din, int base, BackboneP& bp) {
  int idx = base;
  for (int i = 0; i < 8; ++i) {
    bp.c1[i] = {(const float*)din[idx], (const float*)din[idx+1], (const float*)din[idx+2]}; idx += 3;
    bp.c2[i] = {(const float*)din[idx], (const float*)din[idx+1], (const float*)din[idx+2]}; idx += 3;
    if (HASIDN[i]) {
      bp.idn[i] = {(const float*)din[idx], (const float*)din[idx+1], (const float*)din[idx+2]}; idx += 3;
    } else bp.idn[i] = {nullptr, nullptr, nullptr};
  }
  bp.stem = {(const float*)din[idx], (const float*)din[idx+1], (const float*)din[idx+2]}; idx += 3;
  return idx;
}

static inline int cdiv(int a, int b) { return (a + b - 1) / b; }
static inline int pad_up(int a, int b) { return cdiv(a, b) * b; }

extern "C" void kernel_launch(void* const* d_in, const int* in_sizes, int n_in,
                              void* d_out, int out_size, void* d_ws, size_t ws_size,
                              hipStream_t stream) {
  (void)in_sizes; (void)n_in; (void)out_size;

  // ---- input pointer map (jax tree_flatten order: sorted dict keys) ----
  BackboneP clsP, rpnP;
  int idx = parse_backbone(d_in, 0, clsP);           // cls_params
  LinP cls_lab1 = {(const float*)d_in[idx+0],  (const float*)d_in[idx+1]};
  LinP cls_lab2 = {(const float*)d_in[idx+2],  (const float*)d_in[idx+3]};
  LinP cls_off1 = {(const float*)d_in[idx+4],  (const float*)d_in[idx+5]};
  LinP cls_off2 = {(const float*)d_in[idx+6],  (const float*)d_in[idx+7]};
  LinP rpn_lab1 = {(const float*)d_in[idx+8],  (const float*)d_in[idx+9]};
  LinP rpn_lab2 = {(const float*)d_in[idx+10], (const float*)d_in[idx+11]};
  LinP rpn_off1 = {(const float*)d_in[idx+12], (const float*)d_in[idx+13]};
  LinP rpn_off2 = {(const float*)d_in[idx+14], (const float*)d_in[idx+15]};
  idx += 16;
  idx = parse_backbone(d_in, idx, rpnP);             // rpn_params
  const float* x = (const float*)d_in[idx];          // x [8,3,256,256] f32

  // ---- output regions (flat f32, tuple order) ----
  float* out_rpn_labels  = (float*)d_out;                       // 8*21504*2
  float* out_rpn_offsets = out_rpn_labels  + (long)BATCH*NA_TOT*2;
  float* out_boxes       = out_rpn_offsets + (long)BATCH*NA_TOT*4;
  float* out_cls_labels  = out_boxes       + (long)BATCH*MAX_CAND*4;
  float* out_cls_offsets = out_cls_labels  + (long)BATCH*MAX_CAND*2;
  float* out_face        = out_cls_offsets + (long)BATCH*MAX_CAND*8;

  // ---- workspace layout ----
  char* wsp = (char*)d_ws;
  size_t used = 0;
  auto alloc = [&](size_t bytes) -> void* {
    void* p = wsp + used;
    used += (bytes + 255) & ~(size_t)255;
    return p;
  };
  const long ACT_ELEMS = (long)BATCH * IMG * IMG * 16;          // max activation
  bf16_t* xnhwc  = (bf16_t*)alloc((long)BATCH * IMG * IMG * 3 * 2);
  bf16_t* act[4];
  for (int i = 0; i < 4; ++i) act[i] = (bf16_t*)alloc(ACT_ELEMS * 2);
  bf16_t* wpack  = (bf16_t*)alloc((long)POOLK * 128 * 2);       // biggest repack
  bf16_t* h1     = (bf16_t*)alloc((long)BATCH * FEAT_HW * FEAT_HW * 128 * 2);
  bf16_t* pooled = (bf16_t*)alloc((long)BATCH * MAX_CAND * POOLK * 2);
  bf16_t* fvb    = (bf16_t*)alloc((long)BATCH * MAX_CAND * 128 * 2);
  bf16_t* hoff   = (bf16_t*)alloc((long)BATCH * MAX_CAND * 128 * 2);
  float*  boxes_ws = (float*)alloc((long)BATCH * MAX_CAND * 4 * 4);
  if (used > ws_size) return;  // workspace too small; nothing safe to do

  // ---- helpers ----
  auto conv = [&](const bf16_t* in, int Cin, int H, int W, const ConvP& p,
                  bf16_t* out, int Cout, int ks, int stride, int pad,
                  int relu, const bf16_t* resid) {
    int Cpad = pad_up(Cin, 32), Np = pad_up(Cout, 32);
    long tot = (long)ks * ks * Cpad * Np;   // == swizzled element count
    frcnn_pack_conv<<<(unsigned)cdiv((int)tot, 256), 256, 0, stream>>>(
        p.w, Cout, Cin, Cpad, ks, wpack, Np);
    int Ho = H / stride, Wo = W / stride;
    dim3 grid(cdiv(Ho * Wo, 16), Np / 32, BATCH);
    frcnn_conv_wmma<<<grid, 32, 0, stream>>>(in, Cin, Cpad, H, W, wpack, Np,
                                             p.scale, p.bias, resid,
                                             out, Cout, Ho, Wo, ks, stride, pad, relu);
  };
  auto gemm = [&](const bf16_t* A, int M, int K, const LinP& p, int N,
                  float* outF, bf16_t* outB, int relu) {
    int Np = pad_up(N, 32);
    long tot = (long)K * Np;
    frcnn_pack_lin<<<(unsigned)cdiv((int)tot, 256), 256, 0, stream>>>(p.w, N, K, wpack, Np);
    dim3 grid(M / 16, Np / 32);
    frcnn_gemm_wmma<<<grid, 32, 0, stream>>>(A, M, K, wpack, Np, p.b, N, outF, outB, relu);
  };
  auto backbone = [&](const BackboneP& bp) -> bf16_t* {
    int H = IMG, W = IMG;
    bf16_t* y = act[0];
    conv(xnhwc, 3, H, W, bp.stem, y, 4, 3, 1, 1, 1, nullptr);
    for (int i = 0; i < 8; ++i) {
      int cin = BCFG[i][0], cout = BCFG[i][1], stride = BCFG[i][2];
      auto pick = [&](bf16_t* a, bf16_t* b2, bf16_t* c) -> bf16_t* {
        for (int j = 0; j < 4; ++j)
          if (act[j] != a && act[j] != b2 && act[j] != c) return act[j];
        return act[0];
      };
      bf16_t* t = pick(y, nullptr, nullptr);
      conv(y, cin, H, W, bp.c1[i], t, cout, 3, stride, 1, 1, nullptr);
      int Ho = H / stride, Wo = W / stride;
      bf16_t* r = y;
      if (HASIDN[i]) {
        r = pick(y, t, nullptr);
        conv(y, cin, H, W, bp.idn[i], r, cout, 1, stride, 0, 0, nullptr);
      }
      bf16_t* ynext = pick(y, t, r);
      conv(t, cout, Ho, Wo, bp.c2[i], ynext, cout, 3, 1, 1, 1, r);
      y = ynext; H = Ho; W = Wo;
    }
    return y;  // NHWC bf16 [8,32,32,128] == [8192,128] row-major
  };

  // ---- pipeline ----
  {
    long tot = (long)BATCH * 3 * IMG * IMG;
    frcnn_nchw2nhwc<<<(unsigned)cdiv((int)tot, 256), 256, 0, stream>>>(
        x, xnhwc, BATCH, 3, IMG, IMG);
  }

  // RPN branch
  bf16_t* rpnfeat = backbone(rpnP);
  const int Mfeat = BATCH * FEAT_HW * FEAT_HW;  // 8192
  gemm(rpnfeat, Mfeat, 128, rpn_lab1, 128, nullptr, h1, 1);
  gemm(h1, Mfeat, 128, rpn_lab2, NUM_ANCHOR * 2, out_rpn_labels, nullptr, 0);
  gemm(rpnfeat, Mfeat, 128, rpn_off1, 128, nullptr, h1, 1);
  gemm(h1, Mfeat, 128, rpn_off2, NUM_ANCHOR * 4, out_rpn_offsets, nullptr, 0);

  // candidate selection + box adjust
  frcnn_topk_boxes<<<BATCH, 256, 0, stream>>>(out_rpn_labels, out_rpn_offsets,
                                              out_boxes, boxes_ws);

  // classifier branch
  bf16_t* clsfeat = backbone(clsP);
  frcnn_roi_pool<<<BATCH * MAX_CAND, 256, 0, stream>>>(clsfeat, boxes_ws, pooled);

  const int Mroi = BATCH * MAX_CAND;  // 256
  gemm(pooled, Mroi, POOLK, cls_lab1, 128, out_face, fvb, 1);   // fv + face_vecs
  gemm(fvb, Mroi, 128, cls_lab2, 2, out_cls_labels, nullptr, 0);
  gemm(pooled, Mroi, POOLK, cls_off1, 128, nullptr, hoff, 1);
  gemm(hoff, Mroi, 128, cls_off2, 8, out_cls_offsets, nullptr, 0);
}
